// AttentionPairBias_25305947308312
// MI455X (gfx1250) — compile-verified, matrix operand order checked by reading, not run on hardware
//
#include <hip/hip_runtime.h>
#include <hip/hip_bf16.h>
#include <math.h>

typedef __bf16 bh;
typedef __attribute__((ext_vector_type(16))) __bf16 v16bf;
typedef __attribute__((ext_vector_type(8)))  float  v8f;
typedef unsigned int u32;

#define C_A 768
#define C_S 384
#define C_Z 128
#define HH  16
#define DD  48
#define NN  4096
#define NQC 32
#define NKC 128
#define NBC 128

__device__ __forceinline__ bh f2bf(float f){
  union { float f; u32 u; } x; x.f = f;
  u32 r = x.u + 0x7fffu + ((x.u >> 16) & 1u);
  union { unsigned short s; bh b; } y; y.s = (unsigned short)(r >> 16);
  return y.b;
}
__device__ __forceinline__ float sigmoidf_(float x){ return 1.f / (1.f + expf(-x)); }

// Load one 16x32 bf16 fragment (rows striped lanes 0..15, K split by lane/16,
// K pairs packed per dword) from an LDS tile with even leading dim `ld`.
__device__ __forceinline__ v16bf ld_frag(const bh* base, int ld){
  const int lane = threadIdx.x & 31;
  const int row  = lane & 15;
  const int half = lane >> 4;
  const u32* q = (const u32*)(base + row * ld);
  union { v16bf v; u32 u[8]; } r;
  const int kb = half * 4;
#pragma unroll
  for (int i = 0; i < 4; ++i){ r.u[i] = q[kb + i]; r.u[4 + i] = q[8 + kb + i]; }
  return r.v;
}

__device__ __forceinline__ v8f wmma_bf16(v16bf a, v16bf b, v8f c){
  return __builtin_amdgcn_wmma_f32_16x16x32_bf16(false, a, false, b, (short)0, c, false, false);
}

// CDNA5 async global->LDS copy, 16B per lane (tracked by ASYNCcnt).
__device__ __forceinline__ void async_ld_b128(u32 lds_off, const void* g){
  asm volatile("global_load_async_to_lds_b128 %0, %1, off"
               :: "v"(lds_off), "v"((unsigned long long)(size_t)g) : "memory");
}

// ---------------- weight convert ----------------
__global__ void conv_f32_bf16(const float* __restrict__ s, bh* __restrict__ d, int n){
  for (int i = blockIdx.x * blockDim.x + threadIdx.x; i < n; i += gridDim.x * blockDim.x)
    d[i] = f2bf(__builtin_nontemporal_load(s + i));
}

// ---------------- row prep: LN(s)*gamma, raw s, LN(a) ----------------
__device__ float blk_red(float v, float* buf){
#pragma unroll
  for (int o = 16; o; o >>= 1) v += __shfl_xor(v, o, 32);
  if ((threadIdx.x & 31) == 0) buf[threadIdx.x >> 5] = v;
  __syncthreads();
  float t = buf[0] + buf[1] + buf[2] + buf[3];
  __syncthreads();
  return t;
}

__global__ void __launch_bounds__(128)
prep_rows(const float* __restrict__ a, const float* __restrict__ s,
          const float* __restrict__ gamma_s,
          bh* __restrict__ s_bf, bh* __restrict__ sln_bf, float* __restrict__ lna){
  __shared__ float buf[4];
  const int n = blockIdx.x, tid = threadIdx.x;
  float xs[3]; float sm = 0.f, sq = 0.f;
#pragma unroll
  for (int j = 0; j < 3; ++j){ xs[j] = s[(size_t)n * C_S + tid + 128 * j]; sm += xs[j]; sq += xs[j] * xs[j]; }
  float mean = blk_red(sm, buf) / C_S;
  float var  = blk_red(sq, buf) / C_S - mean * mean;
  float rs = rsqrtf(var + 1e-5f);
#pragma unroll
  for (int j = 0; j < 3; ++j){
    int i = tid + 128 * j;
    s_bf  [(size_t)n * C_S + i] = f2bf(xs[j]);
    sln_bf[(size_t)n * C_S + i] = f2bf((xs[j] - mean) * rs * gamma_s[i]);
  }
  float xa[6]; sm = 0.f; sq = 0.f;
#pragma unroll
  for (int j = 0; j < 6; ++j){ float x = a[(size_t)n * C_A + tid + 128 * j]; xa[j] = x; sm += x; sq += x * x; }
  mean = blk_red(sm, buf) / C_A;
  var  = blk_red(sq, buf) / C_A - mean * mean;
  rs = rsqrtf(var + 1e-5f);
#pragma unroll
  for (int j = 0; j < 6; ++j) lna[(size_t)n * C_A + tid + 128 * j] = (xa[j] - mean) * rs;
}

// ---------------- generic 128x128 bf16 WMMA GEMM, async double-buffered ----------------
// EPI: 0=store f32, 1=sigmoid(acc+bias), 2=(acc+bias)*scale->bf16, 3=acc->bf16,
//      4=extra*(acc+bias)->f32
template <int EPI>
__global__ void __launch_bounds__(256)
gemm_bf16(const bh* __restrict__ A, const bh* __restrict__ B,
          float* __restrict__ outF, bh* __restrict__ outB,
          const float* __restrict__ bias, const float* __restrict__ extra,
          int M, int N, int K, float scale){
  __shared__ bh sA[2][128 * 40];
  __shared__ bh sB[2][128 * 40];
  const int bn = blockIdx.x, bm = blockIdx.y;
  const int tid = threadIdx.x, lane = tid & 31;
  const int wv = tid >> 5, wm = wv >> 1, wn = wv & 1;

  int arow[2], acol[2], bkr[2], bn8[2];
#pragma unroll
  for (int j = 0; j < 2; ++j){
    int ch = tid + 256 * j;
    arow[j] = ch >> 2;  acol[j] = (ch & 3) * 8;   // A tile 128x32, 16B chunks
    bkr[j]  = ch >> 4;  bn8[j]  = (ch & 15) * 8;  // B tile 32x128, 16B chunks
  }
  const u32 ldsA0 = (u32)(size_t)(&sA[0][0]);

  auto issueA = [&](int kt, int bufsel){
#pragma unroll
    for (int j = 0; j < 2; ++j){
      const bh* g = A + (size_t)(bm * 128 + arow[j]) * K + kt * 32 + acol[j];
      u32 loff = ldsA0 + (u32)((bufsel * (128 * 40) + arow[j] * 40 + acol[j]) * 2);
      async_ld_b128(loff, g);
    }
  };
  auto loadB = [&](int kt, uint4* r){
#pragma unroll
    for (int j = 0; j < 2; ++j)
      r[j] = *(const uint4*)(B + (size_t)(kt * 32 + bkr[j]) * N + bn * 128 + bn8[j]);
  };
  auto storeB = [&](int bufsel, const uint4* r){
    unsigned short* dst = (unsigned short*)(&sB[bufsel][0]);
#pragma unroll
    for (int j = 0; j < 2; ++j){
      union { uint4 q; unsigned short s[8]; } u; u.q = r[j];
#pragma unroll
      for (int e = 0; e < 8; ++e) dst[(bn8[j] + e) * 40 + bkr[j]] = u.s[e];
    }
  };

  v8f c[2][4] = {};
  const int KT = K >> 5;
  uint4 br[2], brn[2];
  issueA(0, 0);
  loadB(0, br);
  int buf = 0;
  for (int kt = 0; kt < KT; ++kt){
    const bool more = (kt + 1 < KT);
    if (more){ issueA(kt + 1, buf ^ 1); loadB(kt + 1, brn); }
    storeB(buf, br);                          // reg-staged B transpose into LDS
    if (more) asm volatile("s_wait_asynccnt 2" ::: "memory");  // stage kt A done
    else      asm volatile("s_wait_asynccnt 0" ::: "memory");
    __syncthreads();
    v16bf af[2], bfq[4];
#pragma unroll
    for (int mi = 0; mi < 2; ++mi) af[mi]  = ld_frag(&sA[buf][(wm * 32 + mi * 16) * 40], 40);
#pragma unroll
    for (int ni = 0; ni < 4; ++ni) bfq[ni] = ld_frag(&sB[buf][(wn * 64 + ni * 16) * 40], 40);
#pragma unroll
    for (int mi = 0; mi < 2; ++mi)
#pragma unroll
      for (int ni = 0; ni < 4; ++ni)
        c[mi][ni] = wmma_bf16(af[mi], bfq[ni], c[mi][ni]);
    __syncthreads();
    if (more){ br[0] = brn[0]; br[1] = brn[1]; }
    buf ^= 1;
  }
  const int col = lane & 15, half = lane >> 4;
#pragma unroll
  for (int mi = 0; mi < 2; ++mi)
#pragma unroll
    for (int ni = 0; ni < 4; ++ni)
#pragma unroll
      for (int r = 0; r < 8; ++r){
        int gm = bm * 128 + wm * 32 + mi * 16 + r + half * 8;
        int gn = bn * 128 + wn * 64 + ni * 16 + col;
        float acc = c[mi][ni][r];
        size_t idx = (size_t)gm * N + gn;
        if      (EPI == 0) outF[idx] = acc;
        else if (EPI == 1) outF[idx] = sigmoidf_(acc + bias[gn]);
        else if (EPI == 2) outB[idx] = f2bf((acc + bias[gn]) * scale);
        else if (EPI == 3) outB[idx] = f2bf(acc);
        else               outF[idx] = __builtin_nontemporal_load(extra + idx) * (acc + bias[gn]);
      }
}

// ---------------- a_ln elementwise combine ----------------
__global__ void epi_adaln(const float* __restrict__ t1, const float* __restrict__ t2,
                          const float* __restrict__ lna, const float* __restrict__ bs,
                          bh* __restrict__ aln, int total){
  for (int i = blockIdx.x * blockDim.x + threadIdx.x; i < total; i += gridDim.x * blockDim.x){
    int cn = i % C_A;
    float v1 = __builtin_nontemporal_load(t1 + i);
    float v2 = __builtin_nontemporal_load(t2 + i);
    float vl = __builtin_nontemporal_load(lna + i);
    aln[i] = f2bf(sigmoidf_(v1 + bs[cn]) * vl + v2);
  }
}

// ---------------- bias = LN(z)*gamma @ wz, output [h][w][q][k] ----------------
__global__ void __launch_bounds__(256)
bias_ln_gemm(const float* __restrict__ z, const float* __restrict__ gamma,
             const float* __restrict__ wz, float* __restrict__ biasOut){
  __shared__ bh sLN[128 * 136];
  __shared__ bh sWz[16 * 136];
  const int tid = threadIdx.x, lane = tid & 31, wv = tid >> 5;
  for (int i = tid; i < 128 * 16; i += 256){ int k = i >> 4, n = i & 15; sWz[n * 136 + k] = f2bf(wz[k * 16 + n]); }
  const size_t rowbase = (size_t)blockIdx.x * 128;
  for (int rr = 0; rr < 16; ++rr){
    int row = wv * 16 + rr;
    const float* zr = z + (rowbase + row) * C_Z;
    float x[4], sm = 0.f, sq = 0.f;
#pragma unroll
    for (int j = 0; j < 4; ++j){ x[j] = __builtin_nontemporal_load(zr + lane + 32 * j); sm += x[j]; sq += x[j] * x[j]; }
#pragma unroll
    for (int o = 16; o; o >>= 1){ sm += __shfl_xor(sm, o, 32); sq += __shfl_xor(sq, o, 32); }
    float mean = sm / C_Z, var = sq / C_Z - mean * mean, rs = rsqrtf(var + 1e-5f);
#pragma unroll
    for (int j = 0; j < 4; ++j) sLN[row * 136 + lane + 32 * j] = f2bf((x[j] - mean) * rs * gamma[lane + 32 * j]);
  }
  __syncthreads();
  v8f c = {};
#pragma unroll
  for (int ks = 0; ks < 4; ++ks){
    v16bf afr = ld_frag(&sLN[(wv * 16) * 136 + ks * 32], 136);
    v16bf bfr = ld_frag(&sWz[ks * 32], 136);
    c = wmma_bf16(afr, bfr, c);
  }
  const int col = lane & 15, half = lane >> 4;
#pragma unroll
  for (int r = 0; r < 8; ++r){
    size_t grow = rowbase + wv * 16 + r + half * 8;          // (w*32+q)*128 + k
    int w = (int)(grow >> 12), q = (int)((grow >> 7) & 31), kk = (int)(grow & 127);
    biasOut[((size_t)((col * NBC + w) * NQC + q)) * NKC + kk] = c[r];
  }
}

// ---------------- local attention per (w,h) ----------------
__global__ void __launch_bounds__(128)
attn_kernel(const bh* __restrict__ qs, const bh* __restrict__ ks,
            const bh* __restrict__ vs, const float* __restrict__ biasIn,
            const float* __restrict__ gate, bh* __restrict__ go){
  __shared__ bh    sQ [32 * 72];
  __shared__ bh    sK [128 * 72];
  __shared__ bh    sVT[48 * 136];
  __shared__ float sS [32 * 132];
  __shared__ bh    sP [32 * 136];
  const int w = blockIdx.x, h = blockIdx.y;
  const int tid = threadIdx.x, lane = tid & 31, wv = tid >> 5;
  const bh z0 = f2bf(0.f);
  for (int i = tid; i < 32 * 72;  i += 128) sQ[i] = z0;   // zero K-pad 48..63
  for (int i = tid; i < 128 * 72; i += 128) sK[i] = z0;
  __syncthreads();
  for (int i = tid; i < 32 * 48; i += 128){
    int q = i / 48, d = i % 48;
    sQ[q * 72 + d] = qs[(size_t)(w * 32 + q) * C_A + h * 48 + d];
  }
  const int start = w * 32 - 48;
  for (int i = tid; i < 128 * 48; i += 128){
    int kk = i / 48, d = i % 48;
    int kidx = start + kk; kidx = kidx < 0 ? 0 : (kidx > NN - 1 ? NN - 1 : kidx);
    sK [kk * 72 + d]  = ks[(size_t)kidx * C_A + h * 48 + d];
    sVT[d * 136 + kk] = vs[(size_t)kidx * C_A + h * 48 + d];
  }
  __syncthreads();
  {   // scores = Q K^T (scale folded into q) + bias, masked
    const int mi = wv & 1, nb4 = (wv >> 1) * 4;
    v8f cs[4] = {};
#pragma unroll
    for (int ksb = 0; ksb < 2; ++ksb){
      v16bf af = ld_frag(&sQ[(mi * 16) * 72 + ksb * 32], 72);
#pragma unroll
      for (int j = 0; j < 4; ++j){
        v16bf bf_ = ld_frag(&sK[((nb4 + j) * 16) * 72 + ksb * 32], 72);
        cs[j] = wmma_bf16(af, bf_, cs[j]);
      }
    }
    const int col = lane & 15, half = lane >> 4;
#pragma unroll
    for (int j = 0; j < 4; ++j)
#pragma unroll
      for (int r = 0; r < 8; ++r){
        int qrow = mi * 16 + r + half * 8;
        int kcol = (nb4 + j) * 16 + col;
        int kpos = start + kcol;
        float v = cs[j][r] +
          __builtin_nontemporal_load(biasIn + ((size_t)((h * NBC + w) * NQC + qrow)) * NKC + kcol);
        if (kpos < 0 || kpos >= NN) v = -1e9f;
        sS[qrow * 132 + kcol] = v;
      }
  }
  __syncthreads();
  for (int rr = 0; rr < 8; ++rr){     // softmax over 128 keys, 8 rows/wave
    int q = wv * 8 + rr;
    float x[4], m = -1e30f;
#pragma unroll
    for (int j = 0; j < 4; ++j){ x[j] = sS[q * 132 + lane + 32 * j]; m = fmaxf(m, x[j]); }
#pragma unroll
    for (int o = 16; o; o >>= 1) m = fmaxf(m, __shfl_xor(m, o, 32));
    float sum = 0.f;
#pragma unroll
    for (int j = 0; j < 4; ++j){ x[j] = expf(x[j] - m); sum += x[j]; }
#pragma unroll
    for (int o = 16; o; o >>= 1) sum += __shfl_xor(sum, o, 32);
    float inv = 1.f / sum;
#pragma unroll
    for (int j = 0; j < 4; ++j) sP[q * 136 + lane + 32 * j] = f2bf(x[j] * inv);
  }
  __syncthreads();
  if (wv < 3){                         // O = P V, fused output gate
#pragma unroll
    for (int mi = 0; mi < 2; ++mi){
      v8f c = {};
#pragma unroll
      for (int ksd = 0; ksd < 4; ++ksd){
        v16bf af  = ld_frag(&sP [(mi * 16) * 136 + ksd * 32], 136);
        v16bf bf_ = ld_frag(&sVT[(wv * 16) * 136 + ksd * 32], 136);
        c = wmma_bf16(af, bf_, c);
      }
      const int col = lane & 15, half = lane >> 4;
#pragma unroll
      for (int r = 0; r < 8; ++r){
        int q = mi * 16 + r + half * 8;
        int d = wv * 16 + col;
        size_t idx = (size_t)(w * 32 + q) * C_A + h * 48 + d;
        go[idx] = f2bf(__builtin_nontemporal_load(gate + idx) * c[r]);
      }
    }
  }
}

extern "C" void kernel_launch(void* const* d_in, const int* in_sizes, int n_in,
                              void* d_out, int out_size, void* d_ws, size_t ws_size,
                              hipStream_t stream){
  (void)in_sizes; (void)n_in; (void)out_size; (void)ws_size;
  const float* a     = (const float*)d_in[0];
  const float* s     = (const float*)d_in[1];
  const float* z     = (const float*)d_in[2];
  const float* g_s   = (const float*)d_in[3];
  const float* w_ada = (const float*)d_in[4];
  const float* b_ada = (const float*)d_in[5];
  const float* w_skp = (const float*)d_in[6];
  const float* lnzg  = (const float*)d_in[7];
  const float* wz    = (const float*)d_in[8];
  const float* wq    = (const float*)d_in[9];
  const float* bq    = (const float*)d_in[10];
  const float* wk    = (const float*)d_in[11];
  const float* wvp   = (const float*)d_in[12];
  const float* wg    = (const float*)d_in[13];
  const float* bg    = (const float*)d_in[14];
  const float* wo    = (const float*)d_in[15];
  const float* bo    = (const float*)d_in[16];
  const float* wlast = (const float*)d_in[17];
  const float* blast = (const float*)d_in[18];
  float* out = (float*)d_out;

  char* base = (char*)d_ws; size_t off = 0;
  auto alloc = [&](size_t bytes)->void*{ void* p = base + off; off = (off + bytes + 255) & ~(size_t)255; return p; };

  bh* wAdaB  = (bh*)alloc((size_t)C_S * C_A * 2);
  bh* wSkpB  = (bh*)alloc((size_t)C_S * C_A * 2);
  bh* wLstB  = (bh*)alloc((size_t)C_S * C_A * 2);
  bh* wQB    = (bh*)alloc((size_t)C_A * C_A * 2);
  bh* wKB    = (bh*)alloc((size_t)C_A * C_A * 2);
  bh* wVB    = (bh*)alloc((size_t)C_A * C_A * 2);
  bh* wGB    = (bh*)alloc((size_t)C_A * C_A * 2);
  bh* wOB    = (bh*)alloc((size_t)C_A * C_A * 2);
  bh* wZB    = (bh*)alloc((size_t)C_Z * HH * 2);
  bh* slnB   = (bh*)alloc((size_t)NN * C_S * 2);
  bh* sB_    = (bh*)alloc((size_t)NN * C_S * 2);
  float* lnaF  = (float*)alloc((size_t)NN * C_A * 4);
  float* tmp1F = (float*)alloc((size_t)NN * C_A * 4);
  float* tmp2F = (float*)alloc((size_t)NN * C_A * 4);
  float* glstF = (float*)alloc((size_t)NN * C_A * 4);
  bh* alnB   = (bh*)alloc((size_t)NN * C_A * 2);
  bh* qB     = (bh*)alloc((size_t)NN * C_A * 2);
  bh* kB     = (bh*)alloc((size_t)NN * C_A * 2);
  bh* vB     = (bh*)alloc((size_t)NN * C_A * 2);
  float* gateF = (float*)alloc((size_t)NN * C_A * 4);
  float* biasF = (float*)alloc((size_t)HH * NBC * NQC * NKC * 4);
  bh* goB    = (bh*)alloc((size_t)NN * C_A * 2);

  // 1) weight conversion (one-shot, non-temporal reads)
  conv_f32_bf16<<<512, 256, 0, stream>>>(w_ada, wAdaB, C_S * C_A);
  conv_f32_bf16<<<512, 256, 0, stream>>>(w_skp, wSkpB, C_S * C_A);
  conv_f32_bf16<<<512, 256, 0, stream>>>(wlast, wLstB, C_S * C_A);
  conv_f32_bf16<<<512, 256, 0, stream>>>(wq, wQB, C_A * C_A);
  conv_f32_bf16<<<512, 256, 0, stream>>>(wk, wKB, C_A * C_A);
  conv_f32_bf16<<<512, 256, 0, stream>>>(wvp, wVB, C_A * C_A);
  conv_f32_bf16<<<512, 256, 0, stream>>>(wg, wGB, C_A * C_A);
  conv_f32_bf16<<<512, 256, 0, stream>>>(wo, wOB, C_A * C_A);
  conv_f32_bf16<<<64, 256, 0, stream>>>(wz, wZB, C_Z * HH);

  // 2) row prep
  prep_rows<<<NN, 128, 0, stream>>>(a, s, g_s, sB_, slnB, lnaF);

  // 3) adaLN + last-gate GEMMs
  dim3 gs(C_A / 128, NN / 128);
  gemm_bf16<0><<<gs, 256, 0, stream>>>(slnB, wAdaB, tmp1F, nullptr, nullptr, nullptr, NN, C_A, C_S, 1.f);
  gemm_bf16<0><<<gs, 256, 0, stream>>>(slnB, wSkpB, tmp2F, nullptr, nullptr, nullptr, NN, C_A, C_S, 1.f);
  gemm_bf16<1><<<gs, 256, 0, stream>>>(sB_,  wLstB, glstF, nullptr, blast,  nullptr, NN, C_A, C_S, 1.f);

  // 4) a_ln combine
  epi_adaln<<<3072, 256, 0, stream>>>(tmp1F, tmp2F, lnaF, b_ada, alnB, NN * C_A);

  // 5) projections (scale folded into q)
  const float scale = 1.f / sqrtf((float)DD);
  gemm_bf16<2><<<gs, 256, 0, stream>>>(alnB, wQB, nullptr, qB, bq, nullptr, NN, C_A, C_A, scale);
  gemm_bf16<3><<<gs, 256, 0, stream>>>(alnB, wKB, nullptr, kB, nullptr, nullptr, NN, C_A, C_A, 1.f);
  gemm_bf16<3><<<gs, 256, 0, stream>>>(alnB, wVB, nullptr, vB, nullptr, nullptr, NN, C_A, C_A, 1.f);
  gemm_bf16<1><<<gs, 256, 0, stream>>>(alnB, wGB, gateF, nullptr, bg, nullptr, NN, C_A, C_A, 1.f);

  // 6) pair bias: LN(z)*gamma @ wz  -> [h][w][q][k]  (z streamed NT, 256 MB)
  bias_ln_gemm<<<(NBC * NQC * NKC) / 128, 256, 0, stream>>>(z, lnzg, wz, biasF);

  // 7) local attention with fused output gate
  attn_kernel<<<dim3(NBC, HH), 128, 0, stream>>>(qB, kB, vB, biasF, gateF, goB);

  // 8) output projection with fused +bo and last sigmoid gate
  gemm_bf16<4><<<gs, 256, 0, stream>>>(goB, wOB, out, nullptr, bo, glstF, NN, C_A, C_A, 1.f);
}
